// GRU_53386443489883
// MI455X (gfx1250) — matrix operand order repaired and checked by reading, not compiled, not run on hardware
//
#include <hip/hip_runtime.h>
#include <hip/hip_bf16.h>

// ---------------------------------------------------------------------------
// GRU scan for MI455X (gfx1250, wave32, WMMA).
// Strategy: persistent kernel, weights resident in LDS (147456 B/block),
// bf16 WMMA 16x16x32 with f32 accumulation, one grid-sync per timestep.
// ---------------------------------------------------------------------------

typedef __attribute__((ext_vector_type(16))) __bf16 v16bf;
typedef __attribute__((ext_vector_type(8)))  __bf16 v8bf;
typedef __attribute__((ext_vector_type(8)))  float  v8f;

#define B_DIM 64
#define T_DIM 512
#define D_DIM 512
#define H_DIM 1024

// workspace layout (bytes)
#define OFF_XBF  0ull                       // x in bf16:  B*T*D*2      = 33554432
#define OFF_WTX  33554432ull                // Wx^T bf16:  3*H*D*2      =  3145728
#define OFF_WTH  36700160ull                // Wh^T bf16:  3*H*H*2      =  6291456
#define OFF_HBF  42991616ull                // h double buffer: 2*B*H*2 =   262144
#define OFF_CNT  43253760ull                // 513 step counters
// total ~43.3 MB of d_ws

__device__ __forceinline__ float fast_sigmoid(float x) {
    return 1.0f / (1.0f + __expf(-x));
}
__device__ __forceinline__ float fast_tanh(float x) {
    x = fminf(fmaxf(x, -15.0f), 15.0f);
    float e2 = __expf(2.0f * x);
    return (e2 - 1.0f) / (e2 + 1.0f);
}

__device__ __forceinline__ void grid_sync(unsigned* c, unsigned nblk) {
    __syncthreads();
    if (threadIdx.x == 0) {
        __hip_atomic_fetch_add(c, 1u, __ATOMIC_RELEASE, __HIP_MEMORY_SCOPE_AGENT);
        while (__hip_atomic_load(c, __ATOMIC_ACQUIRE, __HIP_MEMORY_SCOPE_AGENT) < nblk)
            __builtin_amdgcn_s_sleep(1);
    }
    __syncthreads();
}

// ------------------------- prep kernels ------------------------------------

__global__ void k_zero_u32(unsigned* p, int n) {
    int i = blockIdx.x * blockDim.x + threadIdx.x;
    if (i < n) p[i] = 0u;
}

__global__ void k_f32_to_bf16(const float* __restrict__ src,
                              __bf16* __restrict__ dst, size_t n) {
    size_t i = (size_t)blockIdx.x * blockDim.x + threadIdx.x;
    size_t s = (size_t)gridDim.x * blockDim.x;
    for (; i < n; i += s) dst[i] = (__bf16)src[i];
}

// src [K][N] row-major fp32  ->  dst [N][K] bf16 (K-contiguous)
__global__ void k_transpose_bf16(const float* __restrict__ src,
                                 __bf16* __restrict__ dst, int K, int N) {
    size_t total = (size_t)K * N;
    size_t i = (size_t)blockIdx.x * blockDim.x + threadIdx.x;
    size_t s = (size_t)gridDim.x * blockDim.x;
    for (; i < total; i += s) {
        int n = (int)(i / K);
        int k = (int)(i % K);
        dst[i] = (__bf16)src[(size_t)k * N + n];
    }
}

// ------------------------- persistent scan kernel --------------------------
// 64 blocks x 128 threads. Block b owns output columns [16*b, 16*b+16).
// Wave w (0..3) owns rows [16*w, 16*w+16) -> one 16x16 tile per wave, fixed
// for all 512 timesteps. LDS holds this block's Wh (3x16x1024) and Wx
// (3x16x512) slices in K-major (column-of-W contiguous) order.

#define LDS_WH_ELEMS (3 * 16 * 1024)
#define LDS_WX_ELEMS (3 * 16 * 512)
#define SCAN_LDS_BYTES ((LDS_WH_ELEMS + LDS_WX_ELEMS) * 2)   // 147456

__launch_bounds__(128, 1)
__global__ void gru_scan(const __bf16* __restrict__ xbf,   // [B][T][D]
                         const __bf16* __restrict__ wtx,   // [3][H][D]  (z,r,n)
                         const __bf16* __restrict__ wth,   // [3][H][H]  (z,r,n)
                         const float* __restrict__ b_zx, const float* __restrict__ b_zh,
                         const float* __restrict__ b_rx, const float* __restrict__ b_rh,
                         const float* __restrict__ b_nx, const float* __restrict__ b_nh,
                         __bf16* __restrict__ hbf,         // [2][B][H]
                         unsigned* __restrict__ cnt,       // [T+1]
                         float* __restrict__ out)          // [B][H]
{
    extern __shared__ char smem_raw[];
    __bf16* ldsWh = (__bf16*)smem_raw;            // [3][16][1024] K-major
    __bf16* ldsWx = ldsWh + LDS_WH_ELEMS;         // [3][16][512]  K-major

    const int tid  = threadIdx.x;
    const int col0 = blockIdx.x * 16;

    // ---- stage weights into LDS (once; reused for all 512 steps) ----
    {
        uint4* dWh = (uint4*)ldsWh;               // 128 uint4 per (g,n) row
        const uint4* sWh = (const uint4*)wth;
        for (int i = tid; i < 3 * 16 * 128; i += 128) {
            int g = i >> 11;                      // / (16*128)
            int rem = i & 2047;
            int n = rem >> 7;
            int u = rem & 127;
            dWh[i] = sWh[(size_t)(g * H_DIM + col0 + n) * 128 + u];
        }
        uint4* dWx = (uint4*)ldsWx;               // 64 uint4 per (g,n) row
        const uint4* sWx = (const uint4*)wtx;
        for (int i = tid; i < 3 * 16 * 64; i += 128) {
            int g = i / (16 * 64);
            int rem = i % (16 * 64);
            int n = rem >> 6;
            int u = rem & 63;
            dWx[i] = sWx[(size_t)(g * H_DIM + col0 + n) * 64 + u];
        }
    }
    __syncthreads();

    const int wid  = tid >> 5;
    const int lane = tid & 31;
    const int lm   = lane & 15;     // row index within A tile / N index within B tile
    const int lh   = lane >> 4;     // lane half -> selects contiguous K run 16*lh..16*lh+15
    const int mt   = wid;           // M tile (rows 16*mt .. 16*mt+15); rows == batch
    const int arow = mt * 16 + lm;  // row this lane gathers for A operands
    const int col  = col0 + lm;     // output column this lane owns

    // per-column biases, hoisted for the whole scan
    const float cbzx = b_zx[col], cbzh = b_zh[col];
    const float cbrx = b_rx[col], cbrh = b_rh[col];
    const float cbnx = b_nx[col], cbnh = b_nh[col];

    v8f h = {0.f, 0.f, 0.f, 0.f, 0.f, 0.f, 0.f, 0.f};   // h_old tile, registers

    // write h0 = 0 into buffer 0 (this wave's tile region)
    #pragma unroll
    for (int r = 0; r < 8; ++r) {
        int row = mt * 16 + r + 8 * lh;
        hbf[(size_t)row * H_DIM + col] = (__bf16)0.f;
    }
    grid_sync(&cnt[0], gridDim.x);

    for (int t = 0; t < T_DIM; ++t) {
        const int rd = t & 1, wr = rd ^ 1;
        const __bf16* hrd = hbf + (size_t)rd * (B_DIM * H_DIM);
        __bf16*       hwr = hbf + (size_t)wr * (B_DIM * H_DIM);

        v8f az  = {0.f,0.f,0.f,0.f,0.f,0.f,0.f,0.f};
        v8f ar  = {0.f,0.f,0.f,0.f,0.f,0.f,0.f,0.f};
        v8f anx = {0.f,0.f,0.f,0.f,0.f,0.f,0.f,0.f};
        v8f anh = {0.f,0.f,0.f,0.f,0.f,0.f,0.f,0.f};

        // ---- recurrent part: h_{t-1} @ W_{z,r,n}h, K = 1024 ----
        const __bf16* aptr = hrd + (size_t)arow * H_DIM + lh * 16;
        #pragma unroll 4
        for (int kk = 0; kk < 32; ++kk) {
            // A: 16 contiguous K values per lane (K = 16*lh + s), matching B
            v8bf alo = *(const v8bf*)(aptr + kk * 32);
            v8bf ahi = *(const v8bf*)(aptr + kk * 32 + 8);
            v16bf a = __builtin_shufflevector(alo, ahi,
                        0,1,2,3,4,5,6,7,8,9,10,11,12,13,14,15);
            // B: 16 contiguous K values for column lm (K-major LDS)
            const __bf16* bb = ldsWh + (size_t)lm * 1024 + kk * 32 + lh * 16;
            v16bf bz = *(const v16bf*)(bb);
            v16bf br = *(const v16bf*)(bb + 16 * 1024);
            v16bf bn = *(const v16bf*)(bb + 32 * 1024);
            az  = __builtin_amdgcn_wmma_f32_16x16x32_bf16(false, a, false, bz, (short)0, az,  false, false);
            ar  = __builtin_amdgcn_wmma_f32_16x16x32_bf16(false, a, false, br, (short)0, ar,  false, false);
            anh = __builtin_amdgcn_wmma_f32_16x16x32_bf16(false, a, false, bn, (short)0, anh, false, false);
        }

        // ---- input part: x_t @ W_{z,r,n}x, K = 512 (fused, no P buffer) ----
        const __bf16* xptr = xbf + ((size_t)arow * T_DIM + t) * D_DIM + lh * 16;
        #pragma unroll 4
        for (int kk = 0; kk < 16; ++kk) {
            v8bf alo = *(const v8bf*)(xptr + kk * 32);
            v8bf ahi = *(const v8bf*)(xptr + kk * 32 + 8);
            v16bf a = __builtin_shufflevector(alo, ahi,
                        0,1,2,3,4,5,6,7,8,9,10,11,12,13,14,15);
            const __bf16* bb = ldsWx + (size_t)lm * 512 + kk * 32 + lh * 16;
            v16bf bz = *(const v16bf*)(bb);
            v16bf br = *(const v16bf*)(bb + 16 * 512);
            v16bf bn = *(const v16bf*)(bb + 32 * 512);
            az  = __builtin_amdgcn_wmma_f32_16x16x32_bf16(false, a, false, bz, (short)0, az,  false, false);
            ar  = __builtin_amdgcn_wmma_f32_16x16x32_bf16(false, a, false, br, (short)0, ar,  false, false);
            anx = __builtin_amdgcn_wmma_f32_16x16x32_bf16(false, a, false, bn, (short)0, anx, false, false);
        }

        // ---- gates + state update (C/D layout: VGPR r -> M = r + 8*lh) ----
        #pragma unroll
        for (int r = 0; r < 8; ++r) {
            float z  = fast_sigmoid(az[r] + cbzx + cbzh);
            float rg = fast_sigmoid(ar[r] + cbrx + cbrh);
            float nn = fast_tanh(anx[r] + cbnx + rg * (anh[r] + cbnh));
            float hn = (1.0f - z) * nn + z * h[r];
            h[r] = hn;
            int row = mt * 16 + r + 8 * lh;
            hwr[(size_t)row * H_DIM + col] = (__bf16)hn;
        }

        grid_sync(&cnt[t + 1], gridDim.x);
    }

    // final h (fp32) -> d_out [B][H]
    #pragma unroll
    for (int r = 0; r < 8; ++r) {
        int row = mt * 16 + r + 8 * lh;
        out[(size_t)row * H_DIM + col] = h[r];
    }
}

// ---------------------------------------------------------------------------

extern "C" void kernel_launch(void* const* d_in, const int* in_sizes, int n_in,
                              void* d_out, int out_size, void* d_ws, size_t ws_size,
                              hipStream_t stream) {
    (void)in_sizes; (void)n_in; (void)out_size; (void)ws_size;

    const float* x    = (const float*)d_in[0];
    const float* W_zx = (const float*)d_in[1];
    const float* W_zh = (const float*)d_in[2];
    const float* W_rx = (const float*)d_in[3];
    const float* W_rh = (const float*)d_in[4];
    const float* W_nx = (const float*)d_in[5];
    const float* W_nh = (const float*)d_in[6];
    const float* b_zx = (const float*)d_in[7];
    const float* b_zh = (const float*)d_in[8];
    const float* b_rx = (const float*)d_in[9];
    const float* b_rh = (const float*)d_in[10];
    const float* b_nx = (const float*)d_in[11];
    const float* b_nh = (const float*)d_in[12];

    char* ws = (char*)d_ws;
    __bf16*   xbf = (__bf16*)(ws + OFF_XBF);
    __bf16*   wtx = (__bf16*)(ws + OFF_WTX);   // [3][H][D], gate order z,r,n
    __bf16*   wth = (__bf16*)(ws + OFF_WTH);   // [3][H][H], gate order z,r,n
    __bf16*   hbf = (__bf16*)(ws + OFF_HBF);
    unsigned* cnt = (unsigned*)(ws + OFF_CNT);

    // reset sync counters every call (graph replay safe)
    k_zero_u32<<<3, 256, 0, stream>>>(cnt, 513);

    // x -> bf16
    k_f32_to_bf16<<<2048, 256, 0, stream>>>(x, xbf, (size_t)B_DIM * T_DIM * D_DIM);

    // weights -> bf16, transposed to [H][K] so GEMM B-operands are K-contiguous
    k_transpose_bf16<<<512, 256, 0, stream>>>(W_zx, wtx + 0 * (size_t)H_DIM * D_DIM, D_DIM, H_DIM);
    k_transpose_bf16<<<512, 256, 0, stream>>>(W_rx, wtx + 1 * (size_t)H_DIM * D_DIM, D_DIM, H_DIM);
    k_transpose_bf16<<<512, 256, 0, stream>>>(W_nx, wtx + 2 * (size_t)H_DIM * D_DIM, D_DIM, H_DIM);
    k_transpose_bf16<<<1024, 256, 0, stream>>>(W_zh, wth + 0 * (size_t)H_DIM * H_DIM, H_DIM, H_DIM);
    k_transpose_bf16<<<1024, 256, 0, stream>>>(W_rh, wth + 1 * (size_t)H_DIM * H_DIM, H_DIM, H_DIM);
    k_transpose_bf16<<<1024, 256, 0, stream>>>(W_nh, wth + 2 * (size_t)H_DIM * H_DIM, H_DIM, H_DIM);

    // persistent scan: 64 blocks (one per 16 output columns), 4 waves each,
    // 147456 B dynamic LDS for resident weight slices
    gru_scan<<<64, 128, SCAN_LDS_BYTES, stream>>>(
        xbf, wtx, wth,
        b_zx, b_zh, b_rx, b_rh, b_nx, b_nh,
        hbf, cnt, (float*)d_out);
}
